// GCN_10222022164973
// MI455X (gfx1250) — matrix operand order repaired and checked by reading, not compiled, not run on hardware
//
#include <hip/hip_runtime.h>
#include <math.h>

typedef __attribute__((ext_vector_type(16))) _Float16 v16h;
typedef __attribute__((ext_vector_type(8)))  float    v8f;

#define N_NODES 100000
#define N_EDGES 1600000
#define IN_F    512
#define HID     128
#define N_CLS   64

// ---------------------------------------------------------------------------
// WMMA fragment layout used here (wave32): contiguous k = 16*(lane/16) + slot, not the ISA table order.
// A (16x32 f16)  element (m,k): lane = (k>>4)*16 + m, slot = k&15
// B (32x16 f16)  element (k,c): lane = (k>>4)*16 + (c&15), slot = k&15
// C/D (16x16 f32) vgpr r: row = (lane>>4)*8 + r, col = lane&15
// Fragments stored lane-major (16 halves = 32B contiguous per lane) so operand
// loads are single b128 pairs from LDS or global.
// ---------------------------------------------------------------------------
__device__ __forceinline__ void frag_a_addr(int m, int k, int& lane, int& slot) {
  lane = ((k >> 4) << 4) + m;
  slot = k & 15;
}

// LDS byte offset of a __shared__ object = low 32 bits of its generic address
__device__ __forceinline__ unsigned lds_addr_of(const void* p) {
  return (unsigned)(unsigned long long)p;
}

// GLOBAL_LOAD_ASYNC_TO_LDS_B128 (GVS mode): VDST = LDS byte address VGPR,
// VADDR = 32-bit unsigned byte offset, SADDR = 64-bit base. Tracked by ASYNCcnt.
__device__ __forceinline__ void async_g2l_b128(unsigned lds_off, unsigned gvoff,
                                               const void* gbase) {
  asm volatile("global_load_async_to_lds_b128 %0, %1, %2"
               :: "v"(lds_off), "v"(gvoff), "s"(gbase)
               : "memory");
}

__device__ __forceinline__ void wait_async0() {
  asm volatile("s_wait_asynccnt 0x0" ::: "memory");
}

// ============== weight pre-swizzle: f32 row-major -> f16 fragments ==========
// Wf laid out [K/32][N/16][lane 0..31][slot 0..15] so each wave reads its
// 32x16 B fragment as one contiguous 1KB block (32 bytes per lane).
template <int N>
__global__ __launch_bounds__(256) void convw_kernel(const float* __restrict__ W,
                                                    _Float16* __restrict__ Wf) {
  const int e = blockIdx.x * 256 + threadIdx.x;
  const int k = e / N, c = e % N;
  const int kb = k >> 5, r = k & 31;
  const int nt = c >> 4;
  const int l = ((r >> 4) << 4) + (c & 15);
  const int s = r & 15;
  Wf[(size_t)kb * (N / 16) * 512 + nt * 512 + l * 16 + s] = (_Float16)W[e];
}

// ============================ GEMM1: Y = X @ W1 =============================
// grid = N_NODES/16, 8 waves, each wave one 16x16 tile of the 128-wide output.
// A tile (16x32 f32, 2KB) streamed via async global->LDS double buffer issued
// one step ahead by wave 0; all threads convert it to the f16 fragment layout;
// B fragments come straight from the pre-swizzled W1f in global (L2-resident).
__global__ __launch_bounds__(256) void gemm1_kernel(const float* __restrict__ X,
                                                    const _Float16* __restrict__ W1f,
                                                    float* __restrict__ Y) {
  __shared__ float    tileF[2][16 * 32];   // raw f32 A tiles (async destination)
  __shared__ _Float16 fragA[2][512];       // f16 A fragments, lane-major
  const int tid  = threadIdx.x;
  const int lane = tid & 31;
  const int wave = tid >> 5;               // N-tile index 0..7
  const size_t m0 = (size_t)blockIdx.x * 16;
  const float* xbase = X + m0 * IN_F;

  // prologue: issue async copy of tile 0 (2KB = 128 x 16B segs = 4 issues)
  if (wave == 0) {
#pragma unroll
    for (int j = 0; j < 4; ++j) {
      const int seg = lane + j * 32;               // seg = r*8 + part
      const unsigned gvoff = (unsigned)((((seg >> 3) * IN_F) + ((seg & 7) << 2)) * 4);
      async_g2l_b128(lds_addr_of(&tileF[0][0]) + seg * 16, gvoff, xbase);
    }
  }

  v8f acc = {};
  for (int kb = 0; kb < IN_F / 32; ++kb) {
    const int buf = kb & 1;
    if (wave == 0) wait_async0();                  // tile[buf] landed in LDS
    __syncthreads();
    if (wave == 0 && kb + 1 < IN_F / 32) {         // prefetch tile kb+1
#pragma unroll
      for (int j = 0; j < 4; ++j) {
        const int seg = lane + j * 32;
        const unsigned gvoff =
            (unsigned)((((seg >> 3) * IN_F) + (kb + 1) * 32 + ((seg & 7) << 2)) * 4);
        async_g2l_b128(lds_addr_of(&tileF[buf ^ 1][0]) + seg * 16, gvoff, xbase);
      }
    }
    // convert f32 tile -> f16 fragment layout (2 elements per thread)
    for (int i = tid; i < 512; i += 256) {
      const int r = i >> 5, c = i & 31;
      int l, s; frag_a_addr(r, c, l, s);
      fragA[buf][l * 16 + s] = (_Float16)tileF[buf][i];
    }
    __syncthreads();

    const v16h a = *(const v16h*)&fragA[buf][lane * 16];
    const v16h b = *(const v16h*)&W1f[(size_t)kb * 4096 + wave * 512 + lane * 16];
    acc = __builtin_amdgcn_wmma_f32_16x16x32_f16(false, a, false, b,
                                                 (short)0, acc, false, false);
  }
  const int    ccol  = wave * 16 + (lane & 15);
  const size_t rbase = m0 + ((lane >> 4) << 3);
#pragma unroll
  for (int r = 0; r < 8; ++r)
    Y[(rbase + r) * HID + ccol] = acc[r];
}

// ===================== GEMM2: Y = relu(H + b1) @ W2 =========================
// grid = N_NODES/32, 8 waves = 2 M-tiles x 4 N-tiles, K = 128. Bias + ReLU +
// f16 conversion fused into the A staging; B fragments direct from global W2f.
__global__ __launch_bounds__(256) void gemm2_kernel(const float* __restrict__ Hpre,
                                                    const float* __restrict__ b1,
                                                    const _Float16* __restrict__ W2f,
                                                    float* __restrict__ Y) {
  __shared__ _Float16 fragA[2][512];       // two 16x32 A fragments (mt = 0,1)
  const int tid  = threadIdx.x;
  const int lane = tid & 31;
  const int wave = tid >> 5;
  const int mt = wave >> 2, nt = wave & 3;
  const size_t blockm = (size_t)blockIdx.x * 32;

  v8f acc = {};
  for (int k0 = 0; k0 < HID; k0 += 32) {
    for (int i = tid; i < 1024; i += 256) {        // 32 rows x 32 k
      const int r = i >> 5, c = i & 31;
      float v = Hpre[(blockm + r) * HID + (k0 + c)] + b1[k0 + c];
      v = v > 0.f ? v : 0.f;                       // fused ReLU
      int l, s; frag_a_addr(r & 15, c, l, s);
      fragA[r >> 4][l * 16 + s] = (_Float16)v;
    }
    __syncthreads();
    const v16h a = *(const v16h*)&fragA[mt][lane * 16];
    const v16h b = *(const v16h*)&W2f[(size_t)(k0 >> 5) * 2048 + nt * 512 + lane * 16];
    acc = __builtin_amdgcn_wmma_f32_16x16x32_f16(false, a, false, b,
                                                 (short)0, acc, false, false);
    __syncthreads();
  }
  const int    ccol  = nt * 16 + (lane & 15);
  const size_t rbase = blockm + mt * 16 + ((lane >> 4) << 3);
#pragma unroll
  for (int r = 0; r < 8; ++r)
    Y[(rbase + r) * N_CLS + ccol] = acc[r];
}

// ========================= SpMM: out[row] += w*h[col] =======================
// One edge per wave; lane owns D/32 consecutive floats. Gathers hit the 192MB
// L2 (feature matrices are 25-51MB); scatter-add via global f32 atomics.
template <int D>
__global__ __launch_bounds__(256) void spmm_kernel(const long long* __restrict__ ei,
                                                   const float* __restrict__ ew,
                                                   const float* __restrict__ h,
                                                   float* __restrict__ out) {
  const int lane = threadIdx.x & 31;
  const int wave = threadIdx.x >> 5;
  const long long e = (long long)blockIdx.x * 8 + wave;
  if (e >= N_EDGES) return;
  const long long row = ei[e];
  const long long col = ei[N_EDGES + e];
  const float w = ew[e];
  constexpr int V = D / 32;
  const float* __restrict__ src = h   + (size_t)col * D + lane * V;
  float*       __restrict__ dst = out + (size_t)row * D + lane * V;
  float v[V];
#pragma unroll
  for (int i = 0; i < V; ++i) v[i] = src[i];       // coalesces to b128/b64
#pragma unroll
  for (int i = 0; i < V; ++i) atomicAdd(&dst[i], w * v[i]);
}

// ================================ utilities =================================
__global__ void zero_kernel(float* __restrict__ p, size_t n) {
  const size_t i = (size_t)blockIdx.x * blockDim.x + threadIdx.x;
  if (i < n) p[i] = 0.f;
}

// bias + log_softmax over 64 classes; one node per wave, 2 classes per lane.
__global__ __launch_bounds__(256) void softmax_kernel(const float* __restrict__ acc,
                                                      const float* __restrict__ b2,
                                                      float* __restrict__ out) {
  const int lane = threadIdx.x & 31;
  const int wave = threadIdx.x >> 5;
  const long long n = (long long)blockIdx.x * 8 + wave;
  if (n >= N_NODES) return;
  const size_t base = (size_t)n * N_CLS + lane * 2;
  float v0 = acc[base + 0] + b2[lane * 2 + 0];
  float v1 = acc[base + 1] + b2[lane * 2 + 1];
  float m = fmaxf(v0, v1);
#pragma unroll
  for (int off = 16; off >= 1; off >>= 1)
    m = fmaxf(m, __shfl_xor(m, off, 32));
  float s = __expf(v0 - m) + __expf(v1 - m);
#pragma unroll
  for (int off = 16; off >= 1; off >>= 1)
    s += __shfl_xor(s, off, 32);
  const float lse = m + __logf(s);
  out[base + 0] = v0 - lse;
  out[base + 1] = v1 - lse;
}

// ================================ launcher ==================================
extern "C" void kernel_launch(void* const* d_in, const int* in_sizes, int n_in,
                              void* d_out, int out_size, void* d_ws, size_t ws_size,
                              hipStream_t stream) {
  const float*     x  = (const float*)d_in[0];
  const long long* ei = (const long long*)d_in[1];   // int64 [2][E]
  const float*     ew = (const float*)d_in[2];
  const float*     W1 = (const float*)d_in[3];
  const float*     b1 = (const float*)d_in[4];
  const float*     W2 = (const float*)d_in[5];
  const float*     b2 = (const float*)d_in[6];
  float* out = (float*)d_out;

  // workspace layout (floats): A | B | C | W1f(f16) | W2f(f16)
  float* wsA = (float*)d_ws;                         // XW1, later acc2
  float* wsB = wsA + (size_t)N_NODES * HID;          // acc1 (pre-bias h1)
  float* wsC = wsB + (size_t)N_NODES * HID;          // relu(h1) @ W2
  _Float16* W1f = (_Float16*)(wsC + (size_t)N_NODES * N_CLS);
  _Float16* W2f = W1f + (size_t)IN_F * HID;

  const size_t nAB = (size_t)N_NODES * HID;          // 12.8M
  const size_t nC  = (size_t)N_NODES * N_CLS;        // 6.4M

  // 0) pre-swizzle weights into WMMA f16 fragment order (tiny, once per call)
  convw_kernel<HID><<<(IN_F * HID) / 256, 256, 0, stream>>>(W1, W1f);
  convw_kernel<N_CLS><<<(HID * N_CLS) / 256, 256, 0, stream>>>(W2, W2f);
  // 1) XW1 = X @ W1                      (WMMA + async global->LDS pipeline)
  gemm1_kernel<<<N_NODES / 16, 256, 0, stream>>>(x, W1f, wsA);
  // 2) acc1 = 0 ; acc1[row] += w * XW1[col]
  zero_kernel<<<(int)((nAB + 255) / 256), 256, 0, stream>>>(wsB, nAB);
  spmm_kernel<HID><<<N_EDGES / 8, 256, 0, stream>>>(ei, ew, wsA, wsB);
  // 3) C = relu(acc1 + b1) @ W2          (WMMA, fused bias+ReLU)
  gemm2_kernel<<<N_NODES / 32, 256, 0, stream>>>(wsB, b1, W2f, wsC);
  // 4) acc2 = 0 (reuse region A) ; acc2[row] += w * C[col]
  zero_kernel<<<(int)((nC + 255) / 256), 256, 0, stream>>>(wsA, nC);
  spmm_kernel<N_CLS><<<N_EDGES / 8, 256, 0, stream>>>(ei, ew, wsC, wsA);
  // 5) out = log_softmax(acc2 + b2)
  softmax_kernel<<<N_NODES / 8, 256, 0, stream>>>(wsA, b2, out);
}
